// KNRM_70153995813373
// MI455X (gfx1250) — matrix operand order repaired and checked by reading, not compiled, not run on hardware
//
#include <hip/hip_runtime.h>
#include <hip/hip_bf16.h>
#include <math.h>

// ---------------------------------------------------------------------------
// KNRM on MI455X (gfx1250, wave32, WMMA + TDM).
//   K1: gather + L2 normalize -> f16  (qnH [B,32,E], dnH [B,D,E] in d_ws)
//   K2: per-batch: TENSOR_LOAD_TO_LDS stages dn chunk (padded rows),
//       f16 WMMA GEMM (32x512x128) -> LDS mm chunks -> exact-match snap ->
//       21-Gaussian pooling via exp-recurrence -> log1p -> MLP -> out[B]
// ---------------------------------------------------------------------------

typedef __attribute__((ext_vector_type(16))) _Float16 v16h;
typedef __attribute__((ext_vector_type(8)))  _Float16 v8h;
typedef __attribute__((ext_vector_type(4)))  _Float16 v4h;
typedef __attribute__((ext_vector_type(8)))  float    v8f;
typedef __attribute__((ext_vector_type(4)))  unsigned int u32x4;
typedef __attribute__((ext_vector_type(8)))  int      i32x8;
typedef __attribute__((ext_vector_type(4)))  int      i32x4;

#define KB   512   // batch
#define KQ   30    // query tokens
#define KQP  32    // padded query rows
#define KD   512   // doc tokens
#define KE   128   // embedding dim
#define KK   21    // gaussian kernels
#define CHUNK 128  // D columns per chunk
#define SMMS 132   // padded sMM row stride (f32) -> bank shift 4/row
#define SBS  136   // padded sB row stride (f16)  -> 272B = TDM pad 16B/256B

#if defined(__has_builtin)
#if __has_builtin(__builtin_amdgcn_tensor_load_to_lds) && \
    __has_builtin(__builtin_amdgcn_s_wait_tensorcnt)
#define USE_TDM 1
#endif
#endif

// ---------------------------------------------------------------------------
// Kernel 1: one wave per token row. Gather emb row, L2-normalize, store f16.
// ---------------------------------------------------------------------------
__global__ void knrm_gather_norm(const int* __restrict__ query,
                                 const int* __restrict__ doc,
                                 const float* __restrict__ emb,
                                 _Float16* __restrict__ qnH,
                                 _Float16* __restrict__ dnH) {
    const int lane = threadIdx.x & 31;
    const int wave = blockIdx.x * (blockDim.x >> 5) + (threadIdx.x >> 5);
    const int QROWS = KB * KQP;

    int tok;
    _Float16* dst;
    if (wave < QROWS) {
        const int b = wave >> 5;
        const int r = wave & 31;
        tok = (r < KQ) ? query[b * KQ + r] : -1;      // pad rows 30,31 -> zeros
        dst = qnH + (size_t)wave * KE;
    } else {
        const int g = wave - QROWS;                   // [0, B*D)
        if (g >= KB * KD) return;
        tok = doc[g];
        dst = dnH + (size_t)g * KE;
    }

    v4h o = (v4h)(_Float16)0.0f;
    if (tok >= 0) {
        const float4 v = *(const float4*)(emb + (size_t)tok * KE + lane * 4);
        float ss = v.x * v.x + v.y * v.y + v.z * v.z + v.w * v.w;
        #pragma unroll
        for (int m = 16; m >= 1; m >>= 1) ss += __shfl_xor(ss, m, 32);
        const float inv = 1.0f / fmaxf(sqrtf(ss), 1e-12f);   // matches ref eps
        o[0] = (_Float16)(v.x * inv);
        o[1] = (_Float16)(v.y * inv);
        o[2] = (_Float16)(v.z * inv);
        o[3] = (_Float16)(v.w * inv);
    }
    *(v4h*)(dst + lane * 4) = o;
}

// ---------------------------------------------------------------------------
// Kernel 2: one block (8 wave32) per batch element.
// ---------------------------------------------------------------------------
__global__ void __launch_bounds__(256)
knrm_fused(const int* __restrict__ query,
           const int* __restrict__ doc,
           const _Float16* __restrict__ qnH,
           const _Float16* __restrict__ dnH,
           const float* __restrict__ w1, const float* __restrict__ b1,
           const float* __restrict__ w2, const float* __restrict__ b2,
           const float* __restrict__ w3, const float* __restrict__ b3,
           float* __restrict__ out) {
    __shared__ _Float16 sA[KQP * KE];        // 8 KB    query tile
    __shared__ float    sMM[KQP * SMMS];     // 16.5 KB cosine chunk (padded)
#ifdef USE_TDM
    __shared__ _Float16 sB[CHUNK * SBS];     // 34 KB   dn chunk (TDM-padded)
#endif
    __shared__ float    sK[KQP * KK];        // final kernel sums [q][k]
    __shared__ int      sQid[KQ];
    __shared__ float    sKm[KK];

    const int b    = blockIdx.x;
    const int tid  = threadIdx.x;
    const int lane = tid & 31;
    const int w    = tid >> 5;               // wave id 0..7 -> N tile

    // ---- stage A tile (32x128 f16) + query ids ----
    *(v16h*)&sA[tid * 16] = *(const v16h*)&qnH[(size_t)b * KQP * KE + tid * 16];
    if (tid < KQ) sQid[tid] = query[b * KQ + tid];
    __syncthreads();

    // ---- A fragments (16x32, contiguous k = 16*(lane/16) + s, same as B), K-reused across chunks ----
    const int ar = lane & 15;
    const int hb = (lane >> 4) * 16;
    v16h afrag[2][4];
    #pragma unroll
    for (int mt = 0; mt < 2; ++mt) {
        #pragma unroll
        for (int kk = 0; kk < 4; ++kk) {
            const _Float16* p = &sA[(mt * 16 + ar) * KE + kk * 32 + hb];
            v8h lo = *(const v8h*)p;
            v8h hi = *(const v8h*)(p + 8);
            afrag[mt][kk] = __builtin_shufflevector(
                lo, hi, 0,1,2,3,4,5,6,7,8,9,10,11,12,13,14,15);
        }
    }

    // ---- per-thread pooling state: q = tid>>3, 16-col slice per chunk ----
    const int pq = tid >> 3;
    const int pd = (tid & 7) * 16;
    float kacc[KK];
    #pragma unroll
    for (int j = 0; j < KK; ++j) kacc[j] = 0.0f;
    const float MU0 = -0.95f;
    const float G1  = 0.60653066f;           // e^{-1/2}
    const float G2  = 0.36787944f;           // e^{-1}

    for (int c = 0; c < KD / CHUNK; ++c) {
#ifdef USE_TDM
        // ---- TDM: DMA 32KB dn chunk -> LDS, pad 16B per 256B row ----
        if (w == 0) {
            const unsigned long long ga =
                (unsigned long long)(uintptr_t)(dnH + ((size_t)b * KD + c * CHUNK) * KE);
            const unsigned int ldsOff = (unsigned int)(uintptr_t)(&sB[0]);
            u32x4 g0;
            g0[0] = 1u;                                        // count=1
            g0[1] = ldsOff;                                    // lds_addr
            g0[2] = (unsigned int)(ga & 0xFFFFFFFFu);          // global_addr lo
            g0[3] = (unsigned int)((ga >> 32) & 0x01FFFFFFu) | (2u << 30); // +type=2
            const unsigned int TD0 = (CHUNK * KE * 2) / 8;     // 4096 x 8B units
            i32x8 g1;
            g1[0] = (int)((3u << 16)        // data_size = 8B
                        | (1u << 20)        // pad_enable
                        | (5u << 22)        // pad_interval: every 256B
                        | (3u << 25));      // pad_amount: 4 dwords = 16B
            g1[1] = (int)((TD0 & 0xFFFFu) << 16);              // tensor_dim0 lo
            g1[2] = (int)((TD0 >> 16) | (1u << 16));           // td0 hi | tensor_dim1=1
            g1[3] = (int)((TD0 & 0xFFFFu) << 16);              // tile_dim0 = 4096
            g1[4] = 0;                                         // tile_dim1/2 unused
            g1[5] = (int)TD0;                                  // tensor_dim0_stride
            g1[6] = 0;
            g1[7] = 0;
            i32x4 gz4 = {0, 0, 0, 0};
            i32x8 gz8 = {0, 0, 0, 0, 0, 0, 0, 0};
            // clang-23 / therock-10.0 form: (g0, g1, g2, g3, extra, cpol)
            __builtin_amdgcn_tensor_load_to_lds(g0, g1, gz4, gz4, gz8, 0);
            __builtin_amdgcn_s_wait_tensorcnt(0);
        }
        __syncthreads();                      // publish sB to all waves
#endif
        // ---- GEMM: wave w owns columns [c*128 + w*16, +16), both M tiles ----
        v8f acc0 = {};
        v8f acc1 = {};
#ifdef USE_TDM
        const _Float16* bbase = &sB[(w * 16 + (lane & 15)) * SBS + (lane >> 4) * 16];
#else
        const _Float16* bbase =
            dnH + ((size_t)b * KD + c * CHUNK + w * 16 + (lane & 15)) * KE
                + (lane >> 4) * 16;
#endif
        #pragma unroll
        for (int kk = 0; kk < 4; ++kk) {
#ifdef USE_TDM
            v8h blo = *(const v8h*)(bbase + kk * 32);
            v8h bhi = *(const v8h*)(bbase + kk * 32 + 8);
            const v16h bf = __builtin_shufflevector(
                blo, bhi, 0,1,2,3,4,5,6,7,8,9,10,11,12,13,14,15);
#else
            const v16h bf = *(const v16h*)(bbase + kk * 32);
#endif
            acc0 = __builtin_amdgcn_wmma_f32_16x16x32_f16(
                false, afrag[0][kk], false, bf, (short)0, acc0, false, false);
            acc1 = __builtin_amdgcn_wmma_f32_16x16x32_f16(
                false, afrag[1][kk], false, bf, (short)0, acc1, false, false);
        }
        // scatter C tiles (layout: M = i + 8*(lane>=16), N = lane&15)
        const int rb = (lane >> 4) * 8;
        const int cn = w * 16 + (lane & 15);
        #pragma unroll
        for (int i = 0; i < 8; ++i) {
            sMM[(rb + i) * SMMS + cn]      = acc0[i];
            sMM[(16 + rb + i) * SMMS + cn] = acc1[i];
        }
        __syncthreads();

        // ---- exact-match snap (protect the sigma=1e-3 kernel vs f16 error) ----
        for (int p = tid; p < KQ * CHUNK; p += 256) {
            const int q  = p >> 7;
            const int dl = p & (CHUNK - 1);
            if (sQid[q] == doc[(size_t)b * KD + c * CHUNK + dl])
                sMM[q * SMMS + dl] = 1.0f;
        }
        __syncthreads();

        // ---- pooling: each thread owns (q, 16 cols); kernels 0..19 via
        //      recurrence  k_j = k_0 * r^j * g^{j^2}  (exact rewrite) ----
        if (pq < KQ) {
            const float* row = &sMM[pq * SMMS + pd];
            #pragma unroll 2
            for (int d = 0; d < 16; ++d) {
                const float v  = row[d];
                const float dv = v - MU0;
                float t = __expf(-(dv * dv) * 50.0f);   // k_0
                float s = __expf(10.0f * dv) * G1;      // r * e^{-1/2}
                #pragma unroll
                for (int j = 0; j < 20; ++j) {
                    kacc[j] += t;
                    t *= s;
                    s *= G2;
                }
                const float de = v - 1.0f;
                kacc[20] += __expf(-(de * de) * 5.0e5f);
            }
        }
        __syncthreads();                      // sMM/sB reused next chunk
    }

    // ---- reduce the 8 column-slice threads per q (contiguous lanes) ----
    #pragma unroll
    for (int j = 0; j < KK; ++j) {
        float s = kacc[j];
        s += __shfl_xor(s, 1, 32);
        s += __shfl_xor(s, 2, 32);
        s += __shfl_xor(s, 4, 32);
        kacc[j] = s;
    }
    if (pq < KQ) {
        for (int j = (tid & 7); j < KK; j += 8) sK[pq * KK + j] = kacc[j];
    }
    __syncthreads();

    // ---- km[k] = sum_q log1p(ksum[q][k]) ----
    if (tid < KK) {
        float s = 0.0f;
        for (int q = 0; q < KQ; ++q) s += log1pf(sK[q * KK + tid]);
        sKm[tid] = s;
    }
    __syncthreads();

    // ---- tiny MLP 21 -> 10 -> 5 -> 1 ----
    if (tid == 0) {
        float h1[10], h2[5];
        for (int j = 0; j < 10; ++j) {
            float s = b1[j];
            for (int k = 0; k < KK; ++k) s += sKm[k] * w1[k * 10 + j];
            h1[j] = fmaxf(s, 0.0f);
        }
        for (int j = 0; j < 5; ++j) {
            float s = b2[j];
            for (int i = 0; i < 10; ++i) s += h1[i] * w2[i * 5 + j];
            h2[j] = fmaxf(s, 0.0f);
        }
        float o = b3[0];
        for (int i = 0; i < 5; ++i) o += h2[i] * w3[i];
        out[b] = o;
    }
}

// ---------------------------------------------------------------------------
extern "C" void kernel_launch(void* const* d_in, const int* in_sizes, int n_in,
                              void* d_out, int out_size, void* d_ws, size_t ws_size,
                              hipStream_t stream) {
    (void)in_sizes; (void)n_in; (void)out_size; (void)ws_size;
    const int*   query = (const int*)d_in[0];
    const int*   doc   = (const int*)d_in[1];
    const float* emb   = (const float*)d_in[2];
    const float* w1    = (const float*)d_in[3];
    const float* b1    = (const float*)d_in[4];
    const float* w2    = (const float*)d_in[5];
    const float* b2    = (const float*)d_in[6];
    const float* w3    = (const float*)d_in[7];
    const float* b3    = (const float*)d_in[8];
    float*       out   = (float*)d_out;

    _Float16* qnH = (_Float16*)d_ws;                       // 4 MB
    _Float16* dnH = qnH + (size_t)KB * KQP * KE;           // 64 MB

    const int totalWaves = KB * KQP + KB * KD;             // 278528
    knrm_gather_norm<<<totalWaves / 8, 256, 0, stream>>>(query, doc, emb, qnH, dnH);

    knrm_fused<<<KB, 256, 0, stream>>>(query, doc, qnH, dnH,
                                       w1, b1, w2, b2, w3, b3, out);
}
